// SparseMoEBlock_9328668967125
// MI455X (gfx1250) — matrix operand order repaired and checked by reading, not compiled, not run on hardware
//
#include <hip/hip_runtime.h>
#include <hip/hip_bf16.h>

// ---------------------------------------------------------------------------
// Types / helpers
// ---------------------------------------------------------------------------
typedef __bf16 bf16_t;
typedef __attribute__((ext_vector_type(16))) __bf16 v16bf;
typedef __attribute__((ext_vector_type(8)))  __bf16 v8bf;
typedef __attribute__((ext_vector_type(4)))  __bf16 v4bf;
typedef __attribute__((ext_vector_type(8)))  float  v8f;
typedef __attribute__((ext_vector_type(4)))  unsigned v4u;
typedef __attribute__((ext_vector_type(8)))  int      v8i;
typedef __attribute__((ext_vector_type(4)))  int      v4i;

static __device__ __forceinline__ bf16_t f2bf(float f) {
    unsigned u = __builtin_bit_cast(unsigned, f);
    unsigned r = (u + 0x7FFFu + ((u >> 16) & 1u)) >> 16;   // round-to-nearest-even
    unsigned short s = (unsigned short)r;
    return __builtin_bit_cast(bf16_t, s);
}
static __device__ __forceinline__ float bf2f(bf16_t b) {
    unsigned short s = __builtin_bit_cast(unsigned short, b);
    unsigned u = ((unsigned)s) << 16;
    return __builtin_bit_cast(float, u);
}
static __device__ __forceinline__ v16bf mk16(v8bf lo, v8bf hi) {
    v16bf r;
#pragma unroll
    for (int i = 0; i < 8; ++i) { r[i] = lo[i]; r[8 + i] = hi[i]; }
    return r;
}

// gfx1250 async global->LDS copy (ASYNCcnt-tracked), per-lane 16 bytes.
static __device__ __forceinline__ void async_copy_b128(unsigned ldsOff, const void* gaddr) {
    asm volatile("global_load_async_to_lds_b128 %0, %1, off"
                 :: "v"(ldsOff), "v"(gaddr) : "memory");
}
static __device__ __forceinline__ void wait_async0() {
    asm volatile("s_wait_asynccnt 0" ::: "memory");
}

// gfx1250 Tensor Data Mover: 2D tile (tileX x tileY elems, 2B elems) from global
// to LDS at ldsAddr, LDS padded 4 DWORDs every 16 DWORDs (row stride 40 halves).
// D# bitfields per CDNA5 ISA 8.3/8.4; TENSORcnt-tracked. 6-arg builtin (clang-23).
static __device__ __forceinline__ void tdm_load_2d_bf16(unsigned ldsAddr, const void* gptr,
                                                        unsigned tileX, unsigned tileY,
                                                        unsigned tdim0, unsigned tdim1,
                                                        unsigned strideElems) {
    unsigned long long ga = (unsigned long long)(uintptr_t)gptr;
    v4u g0;
    g0[0] = 1u;                                               // count=1 (valid descriptor)
    g0[1] = ldsAddr;                                          // lds_addr (bytes)
    g0[2] = (unsigned)(ga & 0xFFFFFFFFu);                     // global_addr[31:0]
    g0[3] = (unsigned)((ga >> 32) & 0x1FFFFFFu) | 0x80000000u; // global_addr[56:32] | type=2
    v8i g1;
    g1[0] = (int)(0x00010000u                                 // data_size=1 -> 2 bytes
                  | (1u << 20)                                // pad_enable
                  | (3u << 22)                                // pad_interval: every 16 DWORDs
                  | (3u << 25));                              // pad_amount: 4 DWORDs
    g1[1] = (int)((tdim0 & 0xFFFFu) << 16);                   // tensor_dim0[15:0] @bit48
    g1[2] = (int)(((tdim0 >> 16) & 0xFFFFu) | ((tdim1 & 0xFFFFu) << 16));
    g1[3] = (int)(((tdim1 >> 16) & 0xFFFFu) | ((tileX & 0xFFFFu) << 16)); // tile_dim0
    g1[4] = (int)(tileY & 0xFFFFu);                           // tile_dim1 (tile_dim2=0)
    g1[5] = (int)strideElems;                                 // tensor_dim0_stride[31:0]
    g1[6] = 0;
    g1[7] = 0;
    v4i z4 = {0, 0, 0, 0};
    v8i z8 = {0, 0, 0, 0, 0, 0, 0, 0};
    __builtin_amdgcn_tensor_load_to_lds(g0, g1, z4, z4, z8, 0);
}

// Problem constants (match reference shapes exactly)
#define S_TOK 8192
#define DDIM  1024
#define NEXP  8
#define HDIM  4096
#define KCAP  2048

// ---------------------------------------------------------------------------
// 0) zero init
// ---------------------------------------------------------------------------
__global__ void zero_kernel(float* __restrict__ p, int n) {
    int i = blockIdx.x * blockDim.x + threadIdx.x;
    if (i < n) p[i] = 0.0f;
}

// ---------------------------------------------------------------------------
// 1) fp32 -> bf16 convert (vector by 4)
// ---------------------------------------------------------------------------
__global__ void cvt_f32_bf16(const float* __restrict__ in, bf16_t* __restrict__ out, int n4) {
    int i = blockIdx.x * blockDim.x + threadIdx.x;
    if (i < n4) {
        float4 f = ((const float4*)in)[i];
        v4bf o;
        o[0] = f2bf(f.x); o[1] = f2bf(f.y); o[2] = f2bf(f.z); o[3] = f2bf(f.w);
        ((v4bf*)out)[i] = o;
    }
}

// ---------------------------------------------------------------------------
// 1b) transpose + convert: in fp32 [R,C] -> out bf16 [C,R]; blockIdx.z = batch
// ---------------------------------------------------------------------------
__global__ __launch_bounds__(256) void tcvt_kernel(const float* __restrict__ in,
                                                   bf16_t* __restrict__ out,
                                                   int R, int C) {
    __shared__ float t[32][33];
    const size_t mat = (size_t)blockIdx.z * R * C;
    const int r0 = blockIdx.x * 32, c0 = blockIdx.y * 32;
    const int tx = threadIdx.x & 31, ty = threadIdx.x >> 5;   // 32 x 8
#pragma unroll
    for (int i = 0; i < 4; ++i) {
        int r = ty + i * 8;
        t[r][tx] = in[mat + (size_t)(r0 + r) * C + c0 + tx];
    }
    __syncthreads();
#pragma unroll
    for (int i = 0; i < 4; ++i) {
        int r = ty + i * 8;   // output row = input col
        out[mat + (size_t)(c0 + r) * R + r0 + tx] = f2bf(t[tx][r]);
    }
}

// ---------------------------------------------------------------------------
// 2) Router: scoresT[e*S + tok] = softmax_e(x[tok] . gate_w[e]); wave per token
// ---------------------------------------------------------------------------
__global__ __launch_bounds__(256) void router_kernel(const float* __restrict__ x,
                                                     const float* __restrict__ gw,
                                                     float* __restrict__ scoresT) {
    int tok  = blockIdx.x * 8 + (threadIdx.x >> 5);
    int lane = threadIdx.x & 31;
    float acc[NEXP];
#pragma unroll
    for (int e = 0; e < NEXP; ++e) acc[e] = 0.0f;
    const float* xr = x + (size_t)tok * DDIM;
    for (int d = lane; d < DDIM; d += 32) {
        float xv = xr[d];
#pragma unroll
        for (int e = 0; e < NEXP; ++e) acc[e] += xv * gw[(size_t)e * DDIM + d];
    }
#pragma unroll
    for (int e = 0; e < NEXP; ++e)
        for (int o = 16; o > 0; o >>= 1) acc[e] += __shfl_xor(acc[e], o, 32);
    if (lane == 0) {
        float mx = acc[0];
#pragma unroll
        for (int e = 1; e < NEXP; ++e) mx = fmaxf(mx, acc[e]);
        float sum = 0.0f;
#pragma unroll
        for (int e = 0; e < NEXP; ++e) { acc[e] = __expf(acc[e] - mx); sum += acc[e]; }
        float inv = 1.0f / sum;
#pragma unroll
        for (int e = 0; e < NEXP; ++e) scoresT[(size_t)e * S_TOK + tok] = acc[e] * inv;
    }
}

// ---------------------------------------------------------------------------
// 3) Per-expert top-K: bitonic sort of 8192 (score,idx) in 64 KB LDS
// ---------------------------------------------------------------------------
__global__ __launch_bounds__(1024) void topk_kernel(const float* __restrict__ scoresT,
                                                    float* __restrict__ gating,
                                                    int*   __restrict__ idx,
                                                    float* __restrict__ ones) {
    __shared__ float sv[S_TOK];
    __shared__ int   si[S_TOK];
    const int e = blockIdx.x, tid = threadIdx.x;
    for (int t = tid; t < S_TOK; t += 1024) { sv[t] = scoresT[(size_t)e * S_TOK + t]; si[t] = t; }
    __syncthreads();
    for (int k = 2; k <= S_TOK; k <<= 1) {
        for (int j = k >> 1; j > 0; j >>= 1) {
            for (int i = tid; i < S_TOK; i += 1024) {
                int l = i ^ j;
                if (l > i) {
                    bool head = ((i & k) == 0);  // descending overall
                    float vi = sv[i], vl = sv[l];
                    if (head ? (vi < vl) : (vi > vl)) {
                        sv[i] = vl; sv[l] = vi;
                        int t = si[i]; si[i] = si[l]; si[l] = t;
                    }
                }
            }
            __syncthreads();
        }
    }
    for (int t = tid; t < KCAP; t += 1024) {
        gating[e * KCAP + t] = sv[t];
        idx[e * KCAP + t]    = si[t];
        ones[(size_t)si[t] * NEXP + e] = 1.0f;
    }
}

// ---------------------------------------------------------------------------
// 4) Tiled bf16 WMMA GEMM:  C[M,N] = act(A[M,K] * BT[N,K]^T + bias)
//    Double-buffered software pipeline:
//      A tile: per-lane global_load_async_to_lds_b128 (ASYNCcnt) [supports gather]
//      B tile: TDM tensor_load_to_lds, wave 0 only (TENSORcnt), LDS-padded to SA
//    128x128x32 tile, 256 thr, 4(M)x2(N) waves -> 2x4 v_wmma_f32_16x16x32_bf16.
// ---------------------------------------------------------------------------
#define BM 128
#define BN 128
#define BKT 32
#define SA 40   // LDS row stride in halves (80 B; TDM pad: 4 DWORDs per 16)

enum { ACT_NONE = 0, ACT_SILU = 1, ACT_GELU = 2 };

template <int ACT, bool SCATTER>
__global__ __launch_bounds__(256) void gemm_bf16_wmma(
    const bf16_t* __restrict__ A, const bf16_t* __restrict__ BT,
    const float* __restrict__ bias, int M, int N, int Kd,
    const int* __restrict__ rowIdx,          // optional gather of A rows
    bf16_t* __restrict__ outBf,              // !SCATTER: bf16 [M,N]
    float*  __restrict__ outY,               // SCATTER: y [S,N] fp32
    const float* __restrict__ gatingRow,     // SCATTER: per-row scale [M]
    const int*   __restrict__ scatterIdx) {  // SCATTER: per-row token [M]
    __shared__ bf16_t As[2][BM * SA];
    __shared__ bf16_t Bs[2][BN * SA];
    const int tid = threadIdx.x;
    const int lane = tid & 31, wid = tid >> 5;
    const int waveM = (wid & 3) * 32;        // 4 waves along M
    const int waveN = (wid >> 2) * 64;       // 2 waves along N
    const int bm = blockIdx.x * BM;
    const int bn = blockIdx.y * BN;
    const int hl = lane >> 4;                // half-wave select
    const int l15 = lane & 15;

    const unsigned asBase = (unsigned)(uintptr_t)(void*)&As[0][0];
    const unsigned bsBase = (unsigned)(uintptr_t)(void*)&Bs[0][0];
    const unsigned bufA = (unsigned)(BM * SA * 2);   // bytes per A buffer
    const unsigned bufB = (unsigned)(BN * SA * 2);

    // per-thread A staging coords: row r, 16-half (32 B) chunk at c
    const int r = tid >> 1;
    const int c = (tid & 1) * 16;
    const int arow = rowIdx ? rowIdx[bm + r] : (bm + r);
    const bf16_t* aSrc = A + (size_t)arow * Kd + c;
    const unsigned aOff = (unsigned)(r * SA + c) * 2u;
    const bf16_t* bTileBase = BT + (size_t)bn * Kd;

    const int nk = Kd / BKT;

    // ---- stage 0
    {
        async_copy_b128(asBase + aOff,       aSrc);
        async_copy_b128(asBase + aOff + 16u, aSrc + 8);
        if (tid < 32)
            tdm_load_2d_bf16(bsBase, bTileBase, BKT, BN,
                             (unsigned)Kd, (unsigned)N, (unsigned)Kd);
    }
    if (tid < 32) __builtin_amdgcn_s_wait_tensorcnt(0);
    wait_async0();
    __syncthreads();

    v8f acc[2][4] = {};

    for (int ik = 0; ik < nk; ++ik) {
        const int cur = ik & 1;
        // ---- prefetch stage ik+1 into the other buffer (overlaps with WMMA)
        if (ik + 1 < nk) {
            const int k0n = (ik + 1) * BKT;
            const unsigned nb = (unsigned)(cur ^ 1);
            async_copy_b128(asBase + nb * bufA + aOff,       aSrc + k0n);
            async_copy_b128(asBase + nb * bufA + aOff + 16u, aSrc + k0n + 8);
            if (tid < 32)
                tdm_load_2d_bf16(bsBase + nb * bufB, bTileBase + k0n, BKT, BN,
                                 (unsigned)(Kd - k0n), (unsigned)N, (unsigned)Kd);
        }

        // ---- compute on buffer `cur` (ISA 7.12.2 bf16 fragment layouts)
        const bf16_t* Ac = &As[cur][0];
        const bf16_t* Bc = &Bs[cur][0];
        const int kbA = hl * 8;
        const int kbB = hl * 8;
        v16bf a[2], b[4];
#pragma unroll
        for (int i = 0; i < 2; ++i) {
            int row = waveM + i * 16 + l15;
            v8bf lo = *(const v8bf*)&Ac[row * SA + kbA];
            v8bf hi = *(const v8bf*)&Ac[row * SA + kbA + 16];
            a[i] = mk16(lo, hi);
        }
#pragma unroll
        for (int j = 0; j < 4; ++j) {
            int col = waveN + j * 16 + l15;
            v8bf lo = *(const v8bf*)&Bc[col * SA + kbB];
            v8bf hi = *(const v8bf*)&Bc[col * SA + kbB + 16];
            b[j] = mk16(lo, hi);
        }
#pragma unroll
        for (int i = 0; i < 2; ++i)
#pragma unroll
            for (int j = 0; j < 4; ++j)
                acc[i][j] = __builtin_amdgcn_wmma_f32_16x16x32_bf16(
                    false, a[i], false, b[j], (short)0, acc[i][j], false, false);

        // ---- commit stage ik+1 (all waves' LDS reads of next buffer finished
        //      no later than the previous barrier, so overwrite was safe)
        if (ik + 1 < nk) {
            if (tid < 32) __builtin_amdgcn_s_wait_tensorcnt(0);
            wait_async0();
            __syncthreads();
        }
    }

    // --- epilogue
#pragma unroll
    for (int i = 0; i < 2; ++i)
#pragma unroll
        for (int j = 0; j < 4; ++j) {
            int col = bn + waveN + j * 16 + l15;
            float bcol = bias ? bias[col] : 0.0f;
#pragma unroll
            for (int rr = 0; rr < 8; ++rr) {
                int grow = bm + waveM + i * 16 + rr + 8 * hl;   // C layout: VGPR r -> M=r / r+8
                float v = acc[i][j][rr] + bcol;
                if (ACT == ACT_SILU) {
                    v = v / (1.0f + __expf(-v));
                } else if (ACT == ACT_GELU) {
                    float t = 0.7978845608028654f * (v + 0.044715f * v * v * v);
                    v = 0.5f * v * (1.0f + tanhf(t));
                }
                if (!SCATTER) {
                    outBf[(size_t)grow * N + col] = f2bf(v);
                } else {
                    float g  = gatingRow[grow];
                    int  tok = scatterIdx[grow];
                    atomicAdd(&outY[(size_t)tok * N + col], g * v);
                }
            }
        }
}

// ---------------------------------------------------------------------------
// 5) capacity head: cap = hidden(bf16) @ cap_w2[D,8] + b2  (wave per token)
// ---------------------------------------------------------------------------
__global__ __launch_bounds__(256) void cap2_kernel(const bf16_t* __restrict__ h,
                                                   const float* __restrict__ w2,
                                                   const float* __restrict__ b2,
                                                   float* __restrict__ capOut) {
    int tok  = blockIdx.x * 8 + (threadIdx.x >> 5);
    int lane = threadIdx.x & 31;
    float acc[NEXP];
#pragma unroll
    for (int e = 0; e < NEXP; ++e) acc[e] = 0.0f;
    const bf16_t* hr = h + (size_t)tok * DDIM;
    for (int d = lane; d < DDIM; d += 32) {
        float hv = bf2f(hr[d]);
#pragma unroll
        for (int e = 0; e < NEXP; ++e) acc[e] += hv * w2[(size_t)d * NEXP + e];
    }
#pragma unroll
    for (int e = 0; e < NEXP; ++e)
        for (int o = 16; o > 0; o >>= 1) acc[e] += __shfl_xor(acc[e], o, 32);
    if (lane == 0)
#pragma unroll
        for (int e = 0; e < NEXP; ++e) capOut[(size_t)tok * NEXP + e] = acc[e] + b2[e];
}

// ---------------------------------------------------------------------------
// Launch
// ---------------------------------------------------------------------------
extern "C" void kernel_launch(void* const* d_in, const int* in_sizes, int n_in,
                              void* d_out, int out_size, void* d_ws, size_t ws_size,
                              hipStream_t stream) {
    const float* x     = (const float*)d_in[0];
    const float* gw    = (const float*)d_in[1];
    const float* capW1 = (const float*)d_in[2];
    const float* capB1 = (const float*)d_in[3];
    const float* capW2 = (const float*)d_in[4];
    const float* capB2 = (const float*)d_in[5];
    const float* expW1 = (const float*)d_in[6];
    const float* expB1 = (const float*)d_in[7];
    const float* expW2 = (const float*)d_in[8];
    const float* expB2 = (const float*)d_in[9];

    float* y    = (float*)d_out;                       // [S, D]
    float* ones = y + (size_t)S_TOK * DDIM;            // [S, E]
    float* cap  = ones + (size_t)S_TOK * NEXP;         // [S, E]

    char* w = (char*)d_ws;
    bf16_t* x_bf   = (bf16_t*)w;  w += (size_t)S_TOK * DDIM * sizeof(bf16_t);
    bf16_t* hbuf   = (bf16_t*)w;  w += (size_t)8388608 * sizeof(bf16_t);       // max(S*D, K*H)
    bf16_t* capW1T = (bf16_t*)w;  w += (size_t)DDIM * DDIM * sizeof(bf16_t);   // [D,D]
    bf16_t* w1T    = (bf16_t*)w;  w += (size_t)NEXP * DDIM * HDIM * sizeof(bf16_t); // [E,H,D]
    bf16_t* w2T    = (bf16_t*)w;  w += (size_t)NEXP * HDIM * DDIM * sizeof(bf16_t); // [E,D,H]
    float*  scoresT = (float*)w;  w += (size_t)NEXP * S_TOK * sizeof(float);
    float*  gating  = (float*)w;  w += (size_t)NEXP * KCAP * sizeof(float);
    int*    idxBuf  = (int*)w;    w += (size_t)NEXP * KCAP * sizeof(int);

    // 0) zero y + ones (cap fully overwritten later)
    {
        int n = S_TOK * DDIM + S_TOK * NEXP;
        zero_kernel<<<(n + 255) / 256, 256, 0, stream>>>(y, n);
    }
    // 1) activations -> bf16; weights -> transposed bf16 [N,K]
    {
        int n4 = S_TOK * DDIM / 4;
        cvt_f32_bf16<<<(n4 + 255) / 256, 256, 0, stream>>>(x, x_bf, n4);
        tcvt_kernel<<<dim3(DDIM / 32, DDIM / 32, 1),    256, 0, stream>>>(capW1, capW1T, DDIM, DDIM);
        tcvt_kernel<<<dim3(DDIM / 32, HDIM / 32, NEXP), 256, 0, stream>>>(expW1, w1T, DDIM, HDIM);
        tcvt_kernel<<<dim3(HDIM / 32, DDIM / 32, NEXP), 256, 0, stream>>>(expW2, w2T, HDIM, DDIM);
    }
    // 2) router softmax -> scoresT
    router_kernel<<<S_TOK / 8, 256, 0, stream>>>(x, gw, scoresT);
    // 3) per-expert top-K (+ ones)
    topk_kernel<<<NEXP, 1024, 0, stream>>>(scoresT, gating, idxBuf, ones);
    // 4) capacity predictor GEMM1 (SiLU) -> hbuf, then small head GEMM2 -> cap
    {
        dim3 grid(S_TOK / BM, DDIM / BN);
        gemm_bf16_wmma<ACT_SILU, false><<<grid, 256, 0, stream>>>(
            x_bf, capW1T, capB1, S_TOK, DDIM, DDIM, nullptr, hbuf, nullptr, nullptr, nullptr);
        cap2_kernel<<<S_TOK / 8, 256, 0, stream>>>(hbuf, capW2, capB2, cap);
    }
    // 5) experts: gathered GEMM1 (GELU) -> hbuf; GEMM2 scaled atomic scatter -> y
    for (int e = 0; e < NEXP; ++e) {
        dim3 g1(KCAP / BM, HDIM / BN);
        gemm_bf16_wmma<ACT_GELU, false><<<g1, 256, 0, stream>>>(
            x_bf, w1T + (size_t)e * HDIM * DDIM, expB1 + (size_t)e * HDIM,
            KCAP, HDIM, DDIM, idxBuf + e * KCAP, hbuf, nullptr, nullptr, nullptr);
        dim3 g2(KCAP / BM, DDIM / BN);
        gemm_bf16_wmma<ACT_NONE, true><<<g2, 256, 0, stream>>>(
            hbuf, w2T + (size_t)e * DDIM * HDIM, expB2 + (size_t)e * DDIM,
            KCAP, DDIM, HDIM, nullptr, nullptr, y,
            gating + e * KCAP, idxBuf + e * KCAP);
    }
}
